// CausalSelfAttention_44358422233404
// MI455X (gfx1250) — compile-verified, matrix operand order checked by reading, not run on hardware
//
#include <hip/hip_runtime.h>

#define BN  4
#define TN  2048
#define CN  1024
#define HN  16
#define HDN 64

typedef __attribute__((ext_vector_type(16))) _Float16 v16h;
typedef __attribute__((ext_vector_type(8)))  _Float16 h8;
typedef __attribute__((ext_vector_type(8)))  float    v8f;
typedef int v4i __attribute__((vector_size(16)));

#if defined(__has_builtin)
#  if __has_builtin(__builtin_amdgcn_global_load_async_to_lds_b128)
#    define HAVE_ASYNC_LDS 1
#  else
#    define HAVE_ASYNC_LDS 0
#  endif
#else
#  define HAVE_ASYNC_LDS 0
#endif

__device__ __forceinline__ v16h cat8(h8 lo, h8 hi) {
  return __builtin_shufflevector(lo, hi, 0,1,2,3,4,5,6,7,8,9,10,11,12,13,14,15);
}

__device__ __forceinline__ v8f wmma_f16(v16h a, v16h b, v8f c) {
  return __builtin_amdgcn_wmma_f32_16x16x32_f16(false, a, false, b, (short)0, c, false, false);
}

// 16-byte global -> LDS copy; async (ASYNCcnt) when toolchain supports it.
__device__ __forceinline__ void copy16_g2l(const _Float16* g, _Float16* l) {
#if HAVE_ASYNC_LDS
  __builtin_amdgcn_global_load_async_to_lds_b128((v4i*)g, (v4i*)l, 0, 0);
#else
  *(h8*)l = *(const h8*)g;
#endif
}

__device__ __forceinline__ void wait_g2l() {
#if HAVE_ASYNC_LDS
#  if __has_builtin(__builtin_amdgcn_s_wait_asynccnt)
  __builtin_amdgcn_s_wait_asynccnt(0);
#  else
  asm volatile("s_wait_asynccnt 0x0" ::: "memory");
#  endif
#endif
}

// wave-internal LDS ordering: LDS is in-order within a wave; just stop the
// compiler from reordering around this point.
__device__ __forceinline__ void wave_lds_fence() { asm volatile("" ::: "memory"); }

// ---------------------------------------------------------------- fp32 -> f16
__global__ void cvt_f32_to_f16(const float* __restrict__ in, _Float16* __restrict__ out, int n) {
  int i = blockIdx.x * blockDim.x + threadIdx.x;
  if (i < n) out[i] = (_Float16)in[i];
}

// ------------------------------------------------- C = A(MxK) * B(NxK)^T GEMM
// Block computes a 128x128 tile; 8 waves in a 4x2 grid, each wave a 32x64
// sub-tile (2 A-frags x 4 B-frags -> 8 WMMAs, 12 ds_load_b128 per k-step).
// Per 32-wide k-step, 256 threads async-stage A/B slices (128x32 f16) into
// double-buffered LDS, overlapping the next stage with the current math.
__global__ __launch_bounds__(256) void gemm_nt(
    const _Float16* __restrict__ A, const _Float16* __restrict__ B,
    float* __restrict__ Cf, _Float16* __restrict__ Ch,
    int M, int N, int K, int outF16) {
  __shared__ __align__(16) _Float16 Abuf[2][128 * 32];
  __shared__ __align__(16) _Float16 Bbuf[2][128 * 32];

  const int tid  = threadIdx.x;
  const int lane = tid & 31;
  const int wave = tid >> 5;
  const int wr   = wave >> 1;          // 0..3 : 32-row strip
  const int wc   = wave & 1;           // 0..1 : 64-col strip
  const int mn   = lane & 15;
  const int g    = lane >> 4;
  const int row0 = blockIdx.x * 128;
  const int col0 = blockIdx.y * 128;

  auto stage = [&](int bufi, int k0) {
#pragma unroll
    for (int i = 0; i < 2; ++i) {
      int c   = tid + 256 * i;        // 512 16B chunks per 128x32 tile
      int row = c >> 2;
      int ko  = (c & 3) * 8;
      copy16_g2l(A + (size_t)(row0 + row) * K + k0 + ko, &Abuf[bufi][row * 32 + ko]);
      copy16_g2l(B + (size_t)(col0 + row) * K + k0 + ko, &Bbuf[bufi][row * 32 + ko]);
    }
  };

  v8f acc[2][4] = {};
  stage(0, 0);
  for (int k0 = 0, it = 0; k0 < K; k0 += 32, ++it) {
    const int cur = it & 1;
    wait_g2l();
    __syncthreads();                      // staged slice visible to all waves
    if (k0 + 32 < K) stage(cur ^ 1, k0 + 32);  // overlap next stage with math

    const _Float16* Ab = &Abuf[cur][0];
    const _Float16* Bb = &Bbuf[cur][0];
    v16h afrag[2], bfrag[4];
#pragma unroll
    for (int i = 0; i < 2; ++i) {
      const _Float16* ar = Ab + (wr * 32 + i * 16 + mn) * 32;
      afrag[i] = cat8(*(const h8*)(ar + g * 8), *(const h8*)(ar + 16 + g * 8));
    }
#pragma unroll
    for (int t = 0; t < 4; ++t) {
      const _Float16* br = Bb + (wc * 64 + t * 16 + mn) * 32;
      bfrag[t] = cat8(*(const h8*)(br + g * 8), *(const h8*)(br + 16 + g * 8));
    }
#pragma unroll
    for (int i = 0; i < 2; ++i)
#pragma unroll
      for (int t = 0; t < 4; ++t)
        acc[i][t] = wmma_f16(afrag[i], bfrag[t], acc[i][t]);
    __syncthreads();                      // everyone done reading `cur`
  }

#pragma unroll
  for (int i = 0; i < 2; ++i)
#pragma unroll
    for (int t = 0; t < 4; ++t)
#pragma unroll
      for (int r = 0; r < 8; ++r) {
        int row = row0 + wr * 32 + i * 16 + r + 8 * g;
        int col = col0 + wc * 64 + t * 16 + mn;
        size_t idx = (size_t)row * N + col;
        if (outF16) Ch[idx] = (_Float16)acc[i][t][r];
        else        Cf[idx] = acc[i][t][r];
      }
}

// ------------------------------------------------------------- head reshaping
// qkv (B,T,3C) f16 -> Q,K (B,H,T,HD);  V transposed -> (B,H,HD,T)
__global__ void split_heads(const _Float16* __restrict__ qkv, _Float16* __restrict__ Qh,
                            _Float16* __restrict__ Kh, _Float16* __restrict__ Vt) {
  size_t i = (size_t)blockIdx.x * blockDim.x + threadIdx.x;
  const size_t total = (size_t)BN * TN * 3 * CN;
  if (i >= total) return;
  int    c3 = (int)(i % (3 * CN));
  size_t bt = i / (3 * CN);
  int t = (int)(bt % TN);
  int b = (int)(bt / TN);
  int which = c3 / CN;
  int c = c3 % CN;
  int h = c / HDN;
  int d = c % HDN;
  _Float16 v = qkv[i];
  size_t bh = (size_t)b * HN + h;
  if (which == 0)      Qh[(bh * TN + t) * HDN + d] = v;
  else if (which == 1) Kh[(bh * TN + t) * HDN + d] = v;
  else                 Vt[(bh * HDN + d) * TN + t] = v;
}

// --------------------------------------------------------------- attention
// Block = 8 waves covering 128 query rows of one (b,h). Per 32-key step all
// 256 threads async-stage the K tile (32x64) and V tile (64x32, HD-major)
// into double-buffered LDS, shared by all 8 waves; each wave runs its own
// 16-query flash-attention step (wave-uniform causal skip keeps EXEC full).
__global__ __launch_bounds__(256) void flash_attn(
    const _Float16* __restrict__ Qh, const _Float16* __restrict__ Kh,
    const _Float16* __restrict__ Vt, _Float16* __restrict__ ctx) {
  __shared__ __align__(16) _Float16 Kbuf[2][32 * 64];    // [key][d]
  __shared__ __align__(16) _Float16 Vbuf[2][64 * 32];    // [d][key]
  __shared__ __align__(16) float    Sbuf[8][16 * 32];
  __shared__ __align__(16) _Float16 Pbuf[8][16 * 32];
  __shared__ float mrow[8][16], lrow[8][16], arw[8][16];

  const int tid  = threadIdx.x;
  const int lane = tid & 31;
  const int wave = tid >> 5;
  const int mn   = lane & 15;
  const int g    = lane >> 4;
  const int h = blockIdx.y, b = blockIdx.z;
  const int qbase = (blockIdx.x * 8 + wave) * 16;
  const size_t bh = (size_t)b * HN + h;

  // Q tile (16 x 64) as two A-fragments (direct global, read once)
  const _Float16* qrow = Qh + (bh * TN + qbase + mn) * (size_t)HDN;
  v16h q0 = cat8(*(const h8*)(qrow + g * 8),      *(const h8*)(qrow + 16 + g * 8));
  v16h q1 = cat8(*(const h8*)(qrow + 32 + g * 8), *(const h8*)(qrow + 48 + g * 8));

  if (lane < 16) { mrow[wave][lane] = -3.0e38f; lrow[wave][lane] = 0.0f; }
  wave_lds_fence();

  v8f o[4] = {};
  const float scale = 0.125f;                       // 1/sqrt(HD)
  const int nj_w   = (qbase + 47) >> 5;             // this wave's causal limit
  const int nj_max = (blockIdx.x * 128 + 112 + 47) >> 5;  // block max (wave 7)

  auto stageKV = [&](int bufi, int j) {
    copy16_g2l(Kh + (bh * TN + j * 32 + (tid >> 3)) * (size_t)HDN + (tid & 7) * 8,
               &Kbuf[bufi][(tid >> 3) * 64 + (tid & 7) * 8]);
    copy16_g2l(Vt + (bh * HDN + (tid >> 2)) * (size_t)TN + j * 32 + (tid & 3) * 8,
               &Vbuf[bufi][(tid >> 2) * 32 + (tid & 3) * 8]);
  };

  stageKV(0, 0);
  for (int j = 0; j < nj_max; ++j) {
    const int cur = j & 1;
    wait_g2l();
    __syncthreads();                                // tile visible to all waves
    if (j + 1 < nj_max) stageKV(cur ^ 1, j + 1);    // overlap next stage

    if (j < nj_w) {                                 // wave-uniform: EXEC stays full
      const _Float16* kr0 = &Kbuf[cur][mn * 64];         // key cols 0..15
      const _Float16* kr1 = &Kbuf[cur][(16 + mn) * 64];  // key cols 16..31
      v16h k00 = cat8(*(const h8*)(kr0 + g * 8),      *(const h8*)(kr0 + 16 + g * 8));
      v16h k01 = cat8(*(const h8*)(kr0 + 32 + g * 8), *(const h8*)(kr0 + 48 + g * 8));
      v16h k10 = cat8(*(const h8*)(kr1 + g * 8),      *(const h8*)(kr1 + 16 + g * 8));
      v16h k11 = cat8(*(const h8*)(kr1 + 32 + g * 8), *(const h8*)(kr1 + 48 + g * 8));
      // V fragments early so their ds latency hides under softmax VALU work
      v16h vfrag[4];
#pragma unroll
      for (int t = 0; t < 4; ++t) {
        const _Float16* vr = &Vbuf[cur][(t * 16 + mn) * 32];
        vfrag[t] = cat8(*(const h8*)(vr + g * 8), *(const h8*)(vr + 16 + g * 8));
      }

      v8f sL = {}, sR = {};
      sL = wmma_f16(q0, k00, sL); sL = wmma_f16(q1, k01, sL);
      sR = wmma_f16(q0, k10, sR); sR = wmma_f16(q1, k11, sR);

#pragma unroll
      for (int r = 0; r < 8; ++r) {
        int row = r + 8 * g;
        int q = qbase + row;
        int k = j * 32 + mn;
        float vL = sL[r] * scale; if (k > q)      vL = -3.0e38f;
        float vR = sR[r] * scale; if (k + 16 > q) vR = -3.0e38f;
        Sbuf[wave][row * 32 + mn]      = vL;
        Sbuf[wave][row * 32 + 16 + mn] = vR;
      }
      wave_lds_fence();

      if (lane < 16) {                              // online softmax, 1 row/lane
        int row = lane;
        float mOld = mrow[wave][row], mNew = mOld;
#pragma unroll
        for (int c2 = 0; c2 < 32; ++c2) mNew = fmaxf(mNew, Sbuf[wave][row * 32 + c2]);
        float alpha = __expf(mOld - mNew);
        float s = 0.f;
#pragma unroll
        for (int c2 = 0; c2 < 32; ++c2) {
          float p = __expf(Sbuf[wave][row * 32 + c2] - mNew);
          s += p;
          Pbuf[wave][row * 32 + c2] = (_Float16)p;
        }
        lrow[wave][row] = lrow[wave][row] * alpha + s;
        mrow[wave][row] = mNew;
        arw[wave][row]  = alpha;
      }
      wave_lds_fence();

      float al[8];
#pragma unroll
      for (int r = 0; r < 8; ++r) al[r] = arw[wave][r + 8 * g];
#pragma unroll
      for (int t = 0; t < 4; ++t)
#pragma unroll
        for (int r = 0; r < 8; ++r) o[t][r] *= al[r];

      v16h pf = cat8(*(const h8*)(&Pbuf[wave][mn * 32 + g * 8]),
                     *(const h8*)(&Pbuf[wave][mn * 32 + 16 + g * 8]));
#pragma unroll
      for (int t = 0; t < 4; ++t)
        o[t] = wmma_f16(pf, vfrag[t], o[t]);
    }
    __syncthreads();                                // K/V tile free for reuse
  }

  float li[8];
#pragma unroll
  for (int r = 0; r < 8; ++r) li[r] = 1.0f / lrow[wave][r + 8 * g];
#pragma unroll
  for (int t = 0; t < 4; ++t)
#pragma unroll
    for (int r = 0; r < 8; ++r) {
      int row = qbase + r + 8 * g;
      int ch  = h * HDN + t * 16 + mn;
      ctx[((size_t)b * TN + row) * CN + ch] = (_Float16)(o[t][r] * li[r]);
    }
}

// ----------------------------------------------------------------- launcher
extern "C" void kernel_launch(void* const* d_in, const int* in_sizes, int n_in,
                              void* d_out, int out_size, void* d_ws, size_t ws_size,
                              hipStream_t stream) {
  (void)in_sizes; (void)n_in; (void)out_size; (void)ws_size;
  const float* x    = (const float*)d_in[0];
  const float* wqkv = (const float*)d_in[1];
  const float* wout = (const float*)d_in[2];
  float* out = (float*)d_out;

  size_t off = 0;
  auto carve = [&](size_t halves) {
    _Float16* p = (_Float16*)((char*)d_ws + off);
    off += ((halves * sizeof(_Float16)) + 255) & ~(size_t)255;
    return p;
  };
  const size_t nx = (size_t)BN * TN * CN;
  _Float16* xh    = carve(nx);
  _Float16* wqkvh = carve((size_t)3 * CN * CN);
  _Float16* wouth = carve((size_t)CN * CN);
  _Float16* qkvh  = carve((size_t)BN * TN * 3 * CN);
  _Float16* Qh    = carve(nx);
  _Float16* Kh    = carve(nx);
  _Float16* Vt    = carve(nx);
  _Float16* ctxh  = carve(nx);

  cvt_f32_to_f16<<<(int)((nx + 255) / 256), 256, 0, stream>>>(x, xh, (int)nx);
  cvt_f32_to_f16<<<(3 * CN * CN + 255) / 256, 256, 0, stream>>>(wqkv, wqkvh, 3 * CN * CN);
  cvt_f32_to_f16<<<(CN * CN + 255) / 256, 256, 0, stream>>>(wout, wouth, CN * CN);

  // qkv = x @ w_qkv^T : M=8192, N=3072, K=1024 (f16 out)
  gemm_nt<<<dim3(BN * TN / 128, (3 * CN) / 128), 256, 0, stream>>>(
      xh, wqkvh, nullptr, qkvh, BN * TN, 3 * CN, CN, 1);

  split_heads<<<(int)((BN * (size_t)TN * 3 * CN) / 256), 256, 0, stream>>>(qkvh, Qh, Kh, Vt);

  // causal flash attention -> ctx (B,T,C) f16
  flash_attn<<<dim3(TN / 128, HN, BN), 256, 0, stream>>>(Qh, Kh, Vt, ctxh);

  // out = ctx @ w_out^T : M=8192, N=1024, K=1024 (f32 out)
  gemm_nt<<<dim3(BN * TN / 128, CN / 128), 256, 0, stream>>>(
      ctxh, wouth, out, nullptr, BN * TN, CN, CN, 0);
}
